// CrossAttention_58093727645899
// MI455X (gfx1250) — matrix operand order repaired and checked by reading, not compiled, not run on hardware
//
#include <hip/hip_runtime.h>
#include <math.h>

// ---------------------------------------------------------------------------
// CrossAttention on MI455X (gfx1250): bf16 WMMA for both GEMMs, f32 softmax.
//   d_out = attn [8192, 8192] f32  ||  output [8192, 512] f32
// Pipeline:
//   k_f32_to_bf16 : Q, K -> bf16 row-major copies (done once, L2-resident)
//   k_vt          : V -> bf16 transposed copy Vt[j][i] (WMMA-B ready)
//   k_scores      : raw scores = Q@K^T + mask*NEG_BIG (bf16 WMMA)
//   k_stats       : one-pass online softmax stats (rowmax, 1/sumexp)
//   k_attn_out    : normalize -> attn (in place) + out = attn@V (bf16 WMMA)
// ---------------------------------------------------------------------------

typedef __attribute__((ext_vector_type(16))) __bf16 v16bf;
typedef __attribute__((ext_vector_type(8)))  __bf16 v8bf;
typedef __attribute__((ext_vector_type(8)))  float  v8f;

#define R_TOT   8192          // b*t rows (4*2048)
#define J_DIM   512
#define I_DIM   8192
#define NEG_BIG (-1.0e9f)

__device__ __forceinline__ __bf16 f2bf(float f) { return (__bf16)f; }

#define SHUF16(lo, hi) __builtin_shufflevector(lo, hi, 0,1,2,3,4,5,6,7,8,9,10,11,12,13,14,15)

// ===========================================================================
// Kernel 0a: elementwise f32 -> bf16 (8 elements / thread, b128-in, b128-out)
// ===========================================================================
__global__ __launch_bounds__(256)
void k_f32_to_bf16(const float* __restrict__ src, __bf16* __restrict__ dst)
{
    const size_t i = ((size_t)blockIdx.x * 256 + threadIdx.x) * 8;
    float4 a = *(const float4*)(src + i);
    float4 b = *(const float4*)(src + i + 4);
    v8bf o;
    o[0] = f2bf(a.x); o[1] = f2bf(a.y); o[2] = f2bf(a.z); o[3] = f2bf(a.w);
    o[4] = f2bf(b.x); o[5] = f2bf(b.y); o[6] = f2bf(b.z); o[7] = f2bf(b.w);
    *(v8bf*)(dst + i) = o;
}

// ===========================================================================
// Kernel 0b: V [8192 i][512 j] f32 -> Vt [512 j][8192 i] bf16 (32x32 LDS tile)
// ===========================================================================
__global__ __launch_bounds__(256)
void k_vt(const float* __restrict__ V, __bf16* __restrict__ Vt)
{
    __shared__ __bf16 tile[32][33];
    const int r  = threadIdx.x >> 3;        // 0..31
    const int c4 = (threadIdx.x & 7) * 4;   // 0,4,..,28
    const int i0 = blockIdx.y * 32;
    const int j0 = blockIdx.x * 32;

    float4 v = *(const float4*)(V + (size_t)(i0 + r) * J_DIM + j0 + c4);
    tile[c4 + 0][r] = f2bf(v.x);
    tile[c4 + 1][r] = f2bf(v.y);
    tile[c4 + 2][r] = f2bf(v.z);
    tile[c4 + 3][r] = f2bf(v.w);
    __syncthreads();

    __bf16* dst = Vt + (size_t)(j0 + r) * I_DIM + i0 + c4;
    dst[0] = tile[r][c4 + 0];
    dst[1] = tile[r][c4 + 1];
    dst[2] = tile[r][c4 + 2];
    dst[3] = tile[r][c4 + 3];
}

// ===========================================================================
// Kernel 1: raw scores = Q @ K^T + mask*NEG_BIG   (written into attn region)
// Tile: 32 rows (t) x 64 cols (i) per 128-thread workgroup (4 waves).
// ===========================================================================
__global__ __launch_bounds__(128)
void k_scores(const __bf16* __restrict__ Kbf, const __bf16* __restrict__ Qbf,
              const float* __restrict__ mask, float* __restrict__ scores)
{
    const int lane  = threadIdx.x & 31;
    const int w     = threadIdx.x >> 5;
    const int tileM = w & 1;
    const int ngrp  = w >> 1;
    const int ml    = lane & 15;   // A-row / B,D-col
    const int kh    = lane >> 4;   // lane half -> K-block

    const int t0 = blockIdx.y * 32;
    const int i0 = blockIdx.x * 64;

    const __bf16* qrow  = Qbf + (size_t)(t0 + tileM * 16 + ml) * J_DIM;
    const __bf16* krow0 = Kbf + (size_t)(i0 + (ngrp * 2 + 0) * 16 + ml) * J_DIM;
    const __bf16* krow1 = Kbf + (size_t)(i0 + (ngrp * 2 + 1) * 16 + ml) * J_DIM;

    // prefetch the streamed (HBM) mask lines for the epilogue
    {
        const int rowg = t0 + tileM * 16 + 8 * kh;
        const int cg0  = i0 + (ngrp * 2) * 16 + ml;
        #pragma unroll
        for (int r = 0; r < 8; ++r)
            __builtin_prefetch(mask + (size_t)(rowg + r) * I_DIM + cg0, 0, 1);
    }

    v8f acc0 = {};
    v8f acc1 = {};

    for (int jc = 0; jc < J_DIM; jc += 32) {
        // A fragment (16x32): lane holds k = kh*16 + {0..15}
        const __bf16* qp = qrow + jc + kh * 16;
        v8bf alo = *(const v8bf*)(qp);
        v8bf ahi = *(const v8bf*)(qp + 8);
        v16bf a  = SHUF16(alo, ahi);

        // B fragments (32x16): lane holds k = kh*16 + {0..15} (32B contiguous)
        v16bf b0 = *(const v16bf*)(krow0 + jc + kh * 16);
        v16bf b1 = *(const v16bf*)(krow1 + jc + kh * 16);

        acc0 = __builtin_amdgcn_wmma_f32_16x16x32_bf16(false, a, false, b0, (short)0, acc0, false, false);
        acc1 = __builtin_amdgcn_wmma_f32_16x16x32_bf16(false, a, false, b1, (short)0, acc1, false, false);
    }

    // D layout: VGPR r -> row m = r + 8*kh, col n = ml
    #pragma unroll
    for (int r = 0; r < 8; ++r) {
        const int rowg = t0 + tileM * 16 + r + 8 * kh;
        const int cg0  = i0 + (ngrp * 2 + 0) * 16 + ml;
        const size_t idx0 = (size_t)rowg * I_DIM + cg0;
        const size_t idx1 = idx0 + 16;
        scores[idx0] = acc0[r] + mask[idx0] * NEG_BIG;
        scores[idx1] = acc1[r] + mask[idx1] * NEG_BIG;
    }
}

// ===========================================================================
// Kernel 2: one-pass online softmax stats: stats[row] = (rowmax, 1/sumexp)
// ===========================================================================
__global__ __launch_bounds__(256)
void k_stats(const float* __restrict__ scores, float2* __restrict__ stats)
{
    const int row = blockIdx.x;
    const int tid = threadIdx.x;
    const float* base = scores + (size_t)row * I_DIM;
    __shared__ float2 red[256];

    float m = -3.4e38f;
    float s = 0.f;
    for (int k = tid; k < I_DIM; k += 256) {
        const float v = base[k];
        if (v > m) { s = s * __expf(m - v) + 1.f; m = v; }
        else       { s += __expf(v - m); }
    }
    red[tid] = make_float2(m, s);
    __syncthreads();
    for (int off = 128; off > 0; off >>= 1) {
        if (tid < off) {
            float2 a = red[tid], b = red[tid + off];
            float  M = fmaxf(a.x, b.x);
            red[tid] = make_float2(M, a.y * __expf(a.x - M) + b.y * __expf(b.x - M));
        }
        __syncthreads();
    }
    if (tid == 0) stats[row] = make_float2(red[0].x, 1.0f / red[0].y);
}

// ===========================================================================
// Kernel 3: normalize scores -> attn (in place) and out = attn @ V.
// Tile: 32 rows x 512 cols per 256-thread workgroup (8 waves).
// A (attn chunk) staged as bf16 in LDS; B comes straight from Vt (L2, 32B/lane).
// ===========================================================================
#define APAD 40   // bf16 row stride (80 B: 16B-aligned, bank-skewed)

__global__ __launch_bounds__(256)
void k_attn_out(const __bf16* __restrict__ Vt, const float2* __restrict__ stats,
                float* __restrict__ attn, float* __restrict__ out)
{
    __shared__ __bf16 As[32][APAD];        // [t_local][i_local]

    const int tid   = threadIdx.x;
    const int lane  = tid & 31;
    const int w     = tid >> 5;            // 0..7
    const int tileM = w & 1;
    const int j0w   = (w >> 1) * 128;      // 128-wide j strip per wave pair
    const int ml    = lane & 15;
    const int kh    = lane >> 4;
    const int t0    = blockIdx.x * 32;

    // staging roles
    const int srow  = tid >> 3;            // 0..31: attn row within tile
    const int sc4   = (tid & 7) * 4;       // 4 i-elements per thread
    const float2 st = stats[t0 + srow];

    // per-wave B base pointers into Vt (one row of Vt per output column)
    const __bf16* vtcol[8];
    #pragma unroll
    for (int n = 0; n < 8; ++n)
        vtcol[n] = Vt + (size_t)(j0w + n * 16 + ml) * I_DIM + kh * 16;

    v8f acc[8] = {};

    for (int ic = 0; ic < I_DIM; ic += 32) {
        // ---- stage: scores -> attn (f32, in place) and bf16 into As ----
        const size_t sidx = (size_t)(t0 + srow) * I_DIM + ic + sc4;
        float4 sv = *(const float4*)(attn + sidx);
        if (ic + 32 < I_DIM) __builtin_prefetch(attn + sidx + 32, 0, 1);
        float4 av;
        av.x = __expf(sv.x - st.x) * st.y;
        av.y = __expf(sv.y - st.x) * st.y;
        av.z = __expf(sv.z - st.x) * st.y;
        av.w = __expf(sv.w - st.x) * st.y;
        *(float4*)(attn + sidx) = av;
        As[srow][sc4 + 0] = f2bf(av.x);
        As[srow][sc4 + 1] = f2bf(av.y);
        As[srow][sc4 + 2] = f2bf(av.z);
        As[srow][sc4 + 3] = f2bf(av.w);
        __syncthreads();

        // ---- consume: A from LDS, B straight from Vt, 8 WMMAs per wave ----
        const __bf16* ap = &As[tileM * 16 + ml][kh * 16];
        v8bf alo = *(const v8bf*)(ap);
        v8bf ahi = *(const v8bf*)(ap + 8);
        v16bf a  = SHUF16(alo, ahi);

        #pragma unroll
        for (int n = 0; n < 8; ++n) {
            v16bf bfrag = *(const v16bf*)(vtcol[n] + ic);
            acc[n] = __builtin_amdgcn_wmma_f32_16x16x32_bf16(false, a, false, bfrag, (short)0, acc[n], false, false);
        }
        __syncthreads();
    }

    // ---- epilogue: D layout r -> row r + 8*kh, col ml ----
    #pragma unroll
    for (int n = 0; n < 8; ++n) {
        #pragma unroll
        for (int r = 0; r < 8; ++r) {
            const int rowg = t0 + tileM * 16 + r + 8 * kh;
            const int colg = j0w + n * 16 + ml;
            out[(size_t)rowg * J_DIM + colg] = acc[n][r];
        }
    }
}

// ===========================================================================
extern "C" void kernel_launch(void* const* d_in, const int* in_sizes, int n_in,
                              void* d_out, int out_size, void* d_ws, size_t ws_size,
                              hipStream_t stream)
{
    (void)in_sizes; (void)n_in; (void)out_size; (void)ws_size;

    const float* KV   = (const float*)d_in[0];   // inputs [8192, 512] (K and V)
    const float* Q    = (const float*)d_in[1];   // query  [4, 2048, 512]
    const float* mask = (const float*)d_in[2];   // weightmask [4, 2048, 8192]

    float* attn = (float*)d_out;                       // [8192, 8192]
    float* out  = attn + (size_t)R_TOT * I_DIM;        // [8192, 512]

    // workspace layout (aligned): stats | Qbf | Kbf | Vt   (~24 MB total)
    char* ws = (char*)d_ws;
    float2* stats = (float2*)ws;                                     //  64 KB
    __bf16* Qbf   = (__bf16*)(ws + (64 << 10));                      //   8 MB
    __bf16* Kbf   = (__bf16*)(ws + (64 << 10) + (8u << 20));         //   8 MB
    __bf16* Vt    = (__bf16*)(ws + (64 << 10) + (16u << 20));        //   8 MB

    const int ncvt = (R_TOT * J_DIM) / (256 * 8);                    // 2048 blocks
    k_f32_to_bf16<<<ncvt, 256, 0, stream>>>(Q,  Qbf);
    k_f32_to_bf16<<<ncvt, 256, 0, stream>>>(KV, Kbf);
    k_vt<<<dim3(J_DIM / 32, I_DIM / 32), 256, 0, stream>>>(KV, Vt);

    dim3 g1(I_DIM / 64, R_TOT / 32);
    k_scores<<<g1, 128, 0, stream>>>(Kbf, Qbf, mask, attn);
    k_stats<<<R_TOT, 256, 0, stream>>>(attn, stats);
    k_attn_out<<<R_TOT / 32, 256, 0, stream>>>(Vt, stats, attn, out);
}
